// EmformerEncoder_4853313044920
// MI455X (gfx1250) — matrix operand order repaired and checked by reading, not compiled, not run on hardware
//
#include <hip/hip_runtime.h>
#include <stdint.h>

// Flip this if the async-to-LDS inline asm is rejected by the assembler.
#ifndef EM_NO_ASYNC
#define EM_ASYNC 1
#else
#define EM_ASYNC 0
#endif

// ---------------------------------------------------------------- types ----
typedef unsigned short u16;
typedef __attribute__((ext_vector_type(16))) __bf16       v16bf;
typedef __attribute__((ext_vector_type(8)))  float        v8f;
typedef __attribute__((ext_vector_type(4)))  unsigned int v4u;
typedef __attribute__((ext_vector_type(4)))  float        v4f;

union BFrag { v4u q[2]; v16bf v; };   // 32B = 16 bf16 halves (one WMMA operand per lane)

// ---------------------------------------------------------------- consts ---
#define BB   4
#define SS   1248
#define TT   1024
#define DD   512
#define HH   8
#define DKK  64
#define FFNN 2048
#define LLAY 4
#define RR   224
#define MM   (BB*SS)                       // 4992 rows
#define WT_PER (3*DD*DD + 2*DD*FFNN)       // bf16 elems of transposed weights per layer
#define BPAD 72                            // padded LDS row stride (halves): bank-conflict free

// ---------------------------------------------------------------- helpers --
__device__ __forceinline__ u16 f2bf(float f) {
  union { float f; uint32_t u; } c; c.f = f;
  uint32_t r = c.u + 0x7FFFu + ((c.u >> 16) & 1u);   // round-to-nearest-even
  return (u16)(r >> 16);
}

__device__ __forceinline__ float half_max(float v) {   // reduce across 16-lane half
#pragma unroll
  for (int m = 1; m < 16; m <<= 1) v = fmaxf(v, __shfl_xor(v, m, 32));
  return v;
}
__device__ __forceinline__ float half_sum(float v) {
#pragma unroll
  for (int m = 1; m < 16; m <<= 1) v += __shfl_xor(v, m, 32);
  return v;
}
__device__ __forceinline__ float wave_sum(float v) {
#pragma unroll
  for (int m = 1; m < 32; m <<= 1) v += __shfl_xor(v, m, 32);
  return v;
}

__device__ __forceinline__ v8f wmma_bf16(const BFrag& a, const BFrag& b, v8f c) {
  return __builtin_amdgcn_wmma_f32_16x16x32_bf16(false, a.v, false, b.v,
                                                 (short)0, c, false, false);
}

#if EM_ASYNC
// CDNA5 async DMA: copy 2x16B per lane directly global -> LDS (ASYNCcnt-tracked).
// IOFFSET is applied to BOTH the global source and the LDS destination
// and our second chunk sits at
// +64B on both sides, so one address pair + offset:64 covers it.
__device__ __forceinline__ void em_async_pair(const u16* gsrc, uint32_t ldsoff) {
  asm volatile("global_load_async_to_lds_b128 %0, %1, off\n\t"
               "global_load_async_to_lds_b128 %0, %1, off offset:64"
               :: "v"(ldsoff), "v"(gsrc) : "memory");
}
__device__ __forceinline__ void em_wait_async0() {
  asm volatile("s_wait_asynccnt 0x0" ::: "memory");
}
#endif

// RIGHT_IDX[c] = 128*(c/32 + 1) + c%32
__device__ __forceinline__ int right_idx(int c) { return 128 * ((c >> 5) + 1) + (c & 31); }

// Emformer attention mask: true = masked out. i=query idx in S, j=key idx in S.
__device__ __forceinline__ bool is_masked(int i, int j, int len) {
  if (j < RR) {                        // key is a hard-copied right-context frame
    bool colv = right_idx(j) < len;
    if (i < RR) return !((i == j) && colv);            // diag block
    int t = i - RR;
    return !(((t >> 7) == (j >> 5)) && colv);          // LEFTM[t, j]
  } else {
    int tk = j - RR;
    if (i < RR) {                                      // left^T block
      bool colv = right_idx(i) < len;
      return !(((tk >> 7) == (i >> 5)) && colv);
    }
    int t = i - RR;
    int ct = t >> 7, ck = tk >> 7;
    bool body = (ck == ct) || (ck == ct - 1);          // chunk + left context
    return !(body && (tk < len) && (t < len));
  }
}

// ---------------------------------------------------------------- kernels --
__global__ void em_lens_kernel(const int* __restrict__ alens, int* lens_i, float* out_tail) {
  int b = threadIdx.x;
  if (b < BB) {
    int v = (alens[b] + 3) >> 2;
    lens_i[b] = v;
    out_tail[b] = (float)v;
  }
}

// transpose fp32 [K,N] -> bf16 [N,K]
__global__ void em_wconv_kernel(const float* __restrict__ src, u16* __restrict__ dst,
                                int K, int N) {
  int idx = blockIdx.x * 256 + threadIdx.x;
  if (idx >= K * N) return;
  int n = idx / K, k = idx - n * K;
  dst[idx] = f2bf(src[(size_t)k * N + n]);
}

// frontend: linear(80->128) + stack-4 subsample + right-context gather -> X[B,S,512]
__global__ void __launch_bounds__(128)
em_frontend_kernel(const float* __restrict__ audio, const float* __restrict__ lw,
                   const float* __restrict__ lb, float* __restrict__ X) {
  int blk = blockIdx.x;                 // b*SS + s
  int b = blk / SS, s = blk - b * SS;
  int t = (s < RR) ? right_idx(s) : (s - RR);
  int c = threadIdx.x;                  // 0..127 output column
  const float* abase = audio + (size_t)b * 80 * 4096 + t * 4;
  float a0 = lb[c], a1 = a0, a2 = a0, a3 = a0;
#pragma unroll 4
  for (int f = 0; f < 80; f++) {
    float w = lw[f * 128 + c];
    v4f av = *(const v4f*)(abase + (size_t)f * 4096);
    a0 += av.x * w; a1 += av.y * w; a2 += av.z * w; a3 += av.w * w;
  }
  float* xr = X + (size_t)blk * DD;
  xr[c] = a0; xr[128 + c] = a1; xr[256 + c] = a2; xr[384 + c] = a3;
}

// LN over D=512: 1 wave/row, writes f32 + bf16
__global__ void __launch_bounds__(256)
em_ln_in_kernel(const float* __restrict__ X, const float* __restrict__ g,
                const float* __restrict__ bt, float* __restrict__ H,
                u16* __restrict__ HB) {
  int r = blockIdx.x * 8 + (threadIdx.x >> 5);
  int lane = threadIdx.x & 31;
  const v4f* p = (const v4f*)(X + (size_t)r * DD + lane * 16);
  float vals[16];
#pragma unroll
  for (int q = 0; q < 4; q++) {
    v4f v = p[q];
    vals[q * 4 + 0] = v.x; vals[q * 4 + 1] = v.y; vals[q * 4 + 2] = v.z; vals[q * 4 + 3] = v.w;
  }
  float s = 0.f, s2 = 0.f;
#pragma unroll
  for (int i = 0; i < 16; i++) { s += vals[i]; s2 += vals[i] * vals[i]; }
  s = wave_sum(s); s2 = wave_sum(s2);
  float mean = s * (1.f / DD);
  float var  = s2 * (1.f / DD) - mean * mean;
  float rs = rsqrtf(var + 1e-5f);
#pragma unroll
  for (int i = 0; i < 16; i++) {
    int c = lane * 16 + i;
    float o = (vals[i] - mean) * rs * g[c] + bt[c];
    H[(size_t)r * DD + c] = o;
    HB[(size_t)r * DD + c] = f2bf(o);
  }
}

// residual add + LN -> (AO f32 in-place, Y bf16)
__global__ void __launch_bounds__(256)
em_ln_res_kernel(float* __restrict__ A, const float* __restrict__ Hsrc,
                 const float* __restrict__ g, const float* __restrict__ bt,
                 u16* __restrict__ Yb) {
  int r = blockIdx.x * 8 + (threadIdx.x >> 5);
  int lane = threadIdx.x & 31;
  float vals[16];
  const v4f* pa = (const v4f*)(A + (size_t)r * DD + lane * 16);
  const v4f* ph = (const v4f*)(Hsrc + (size_t)r * DD + lane * 16);
#pragma unroll
  for (int q = 0; q < 4; q++) {
    v4f a = pa[q], h = ph[q];
    vals[q * 4 + 0] = a.x + h.x; vals[q * 4 + 1] = a.y + h.y;
    vals[q * 4 + 2] = a.z + h.z; vals[q * 4 + 3] = a.w + h.w;
  }
  float s = 0.f, s2 = 0.f;
#pragma unroll
  for (int i = 0; i < 16; i++) { s += vals[i]; s2 += vals[i] * vals[i]; }
  s = wave_sum(s); s2 = wave_sum(s2);
  float mean = s * (1.f / DD);
  float rs = rsqrtf(s2 * (1.f / DD) - mean * mean + 1e-5f);
#pragma unroll
  for (int i = 0; i < 16; i++) {
    int c = lane * 16 + i;
    A[(size_t)r * DD + c] = vals[i];                       // AO kept for later residual
    Yb[(size_t)r * DD + c] = f2bf((vals[i] - mean) * rs * g[c] + bt[c]);
  }
}

// residual add + LN -> X f32 (end of block)
__global__ void __launch_bounds__(256)
em_ln_res2_kernel(const float* __restrict__ Z, const float* __restrict__ AO,
                  const float* __restrict__ g, const float* __restrict__ bt,
                  float* __restrict__ Xo) {
  int r = blockIdx.x * 8 + (threadIdx.x >> 5);
  int lane = threadIdx.x & 31;
  float vals[16];
  const v4f* pz = (const v4f*)(Z + (size_t)r * DD + lane * 16);
  const v4f* pa = (const v4f*)(AO + (size_t)r * DD + lane * 16);
#pragma unroll
  for (int q = 0; q < 4; q++) {
    v4f a = pz[q], h = pa[q];
    vals[q * 4 + 0] = a.x + h.x; vals[q * 4 + 1] = a.y + h.y;
    vals[q * 4 + 2] = a.z + h.z; vals[q * 4 + 3] = a.w + h.w;
  }
  float s = 0.f, s2 = 0.f;
#pragma unroll
  for (int i = 0; i < 16; i++) { s += vals[i]; s2 += vals[i] * vals[i]; }
  s = wave_sum(s); s2 = wave_sum(s2);
  float mean = s * (1.f / DD);
  float rs = rsqrtf(s2 * (1.f / DD) - mean * mean + 1e-5f);
#pragma unroll
  for (int i = 0; i < 16; i++) {
    int c = lane * 16 + i;
    Xo[(size_t)r * DD + c] = (vals[i] - mean) * rs * g[c] + bt[c];
  }
}

// 16 ds_loads (two fragment sets) then 8 WMMAs over one staged 64x64 B tile:
// deep ds queue lets s_wait_dscnt retire progressively under the WMMAs.
#define EM_GEMM_COMPUTE(BSMBUF, A0, A1)                                          \
  {                                                                              \
    BFrag bfrA[4], bfrB[4];                                                      \
    _Pragma("unroll") for (int t = 0; t < 4; t++) {                              \
      const u16* br = &(BSMBUF)[(t * 16 + ln) * BPAD + 16 * g];                  \
      bfrA[t].q[0] = *(const v4u*)(br); bfrA[t].q[1] = *(const v4u*)(br + 8);    \
    }                                                                            \
    _Pragma("unroll") for (int t = 0; t < 4; t++) {                              \
      const u16* br = &(BSMBUF)[(t * 16 + ln) * BPAD + 32 + 16 * g];             \
      bfrB[t].q[0] = *(const v4u*)(br); bfrB[t].q[1] = *(const v4u*)(br + 8);    \
    }                                                                            \
    _Pragma("unroll") for (int t = 0; t < 4; t++)                                \
      acc[t] = wmma_bf16(A0, bfrA[t], acc[t]);                                   \
    _Pragma("unroll") for (int t = 0; t < 4; t++)                                \
      acc[t] = wmma_bf16(A1, bfrB[t], acc[t]);                                   \
  }

#define EM_LOAD_A(F0, F1, KOFF)                                                  \
  F0.q[0] = *(const v4u*)(arow + (KOFF) + 16 * g);                               \
  F0.q[1] = *(const v4u*)(arow + (KOFF) + 8 + 16 * g);                           \
  F1.q[0] = *(const v4u*)(arow + (KOFF) + 32 + 16 * g);                          \
  F1.q[1] = *(const v4u*)(arow + (KOFF) + 40 + 16 * g);

// GEMM v5: 8 waves/WG, 128x64 tile, 64x64 bf16 B stage (padded, conflict-free)
// double-buffered in LDS via global_load_async_to_lds_b128 (ASYNCcnt DMA),
// stage loop unrolled 2x with dedicated A-register sets (no rotation movs).
// mode 0: f32 out, row-major ldc   mode 1: bf16 out, row-major ldc
// mode 2: bf16 out [b,h,s,dk]      mode 3: bf16 out [b,h,dk,s] (V transposed)
__global__ void __launch_bounds__(256)
em_gemm128x64(const u16* __restrict__ A, const u16* __restrict__ Wt,
              const float* __restrict__ bias, void* __restrict__ outp,
              int K, int ldc, int mode) {
  __shared__ __align__(16) u16 Bsm[2][64 * BPAD];
  const int tid  = threadIdx.x;
  const int wave = tid >> 5, lane = tid & 31;
  const int g = lane >> 4, ln = lane & 15;
  const int m0 = blockIdx.x * 128, n0 = blockIdx.y * 64;
  const int m0w = m0 + wave * 16;

  // cooperative B-tile loader: thread -> (col bc, two 16B chunks of its 64-k row)
  const int bc = tid >> 2, bco = (tid & 3) * 8;               // col, halves offset
  const u16* bsrc = Wt + (size_t)(n0 + bc) * K + bco;

  const u16* arow = A + (size_t)(m0w + ln) * K;

  v8f acc[4];
#pragma unroll
  for (int t = 0; t < 4; t++)
#pragma unroll
    for (int v = 0; v < 8; v++) acc[t][v] = 0.f;

#if EM_ASYNC
  const uint32_t l0 = (uint32_t)(uintptr_t)&Bsm[0][bc * BPAD + bco];
  const uint32_t l1 = (uint32_t)(uintptr_t)&Bsm[1][bc * BPAD + bco];

  // prologue: DMA tile k=0 into buf0; load A frags for k=0
  em_async_pair(bsrc, l0);
  BFrag aA0, aA1, aB0, aB1;
  EM_LOAD_A(aA0, aA1, 0)

  for (int k0 = 0; k0 < K; k0 += 128) {      // K is a multiple of 128
    const int kn1 = k0 + 64;
    const int kn2 = (k0 + 128 < K) ? (k0 + 128) : 0;
    // ---- even stage: consume buf0 / tile k0 ----
    em_wait_async0();                        // own DMA portion landed
    __syncthreads();                         // all portions landed; buf1 readers done
    em_async_pair(bsrc + kn1, l1);           // DMA next tile -> buf1
    EM_LOAD_A(aB0, aB1, kn1)
    EM_GEMM_COMPUTE(Bsm[0], aA0, aA1)
    // ---- odd stage: consume buf1 / tile kn1 ----
    em_wait_async0();
    __syncthreads();
    em_async_pair(bsrc + kn2, l0);           // DMA next tile -> buf0 (wraps on last)
    EM_LOAD_A(aA0, aA1, kn2)
    EM_GEMM_COMPUTE(Bsm[1], aB0, aB1)
  }
#else
  // fallback: global->reg->ds_store staging (v3)
  v4u pre0 = *(const v4u*)(bsrc);
  v4u pre1 = *(const v4u*)(bsrc + 32);
  BFrag a0, a1;
  EM_LOAD_A(a0, a1, 0)
  int buf = 0;
  for (int k0 = 0; k0 < K; k0 += 64) {
    u16* bs = &Bsm[buf][bc * BPAD + bco];
    *(v4u*)(bs) = pre0;
    *(v4u*)(bs + 32) = pre1;
    __syncthreads();
    const int k0n = (k0 + 64 < K) ? (k0 + 64) : 0;
    pre0 = *(const v4u*)(bsrc + k0n);
    pre1 = *(const v4u*)(bsrc + k0n + 32);
    BFrag an0, an1;
    EM_LOAD_A(an0, an1, k0n)
    EM_GEMM_COMPUTE(Bsm[buf], a0, a1)
    a0 = an0; a1 = an1;
    buf ^= 1;
  }
#endif

  const int bb = m0w / SS, sbase = m0w - bb * SS;   // 16-row subtile never crosses batch
#pragma unroll
  for (int t = 0; t < 4; t++) {
    int ncol = n0 + t * 16 + ln;
    float bvv = bias[ncol];
#pragma unroll
    for (int v = 0; v < 8; v++) {
      float val = acc[t][v] + bvv;
      int r = m0w + v + 8 * g;
      if (mode == 0) {
        ((float*)outp)[(size_t)r * ldc + ncol] = val;
      } else if (mode == 1) {
        ((u16*)outp)[(size_t)r * ldc + ncol] = f2bf(val);
      } else {
        int s = sbase + v + 8 * g;
        int h = ncol >> 6, dk = ncol & 63;
        if (mode == 2)
          ((u16*)outp)[(((size_t)bb * HH + h) * SS + s) * DKK + dk] = f2bf(val);
        else
          ((u16*)outp)[(((size_t)bb * HH + h) * DKK + dk) * SS + s] = f2bf(val);
      }
    }
  }
}

// Fused flash-attention: 1 wave per (b, h, 16-query tile); online softmax over 32-col steps.
// V fragments for the step are loaded before the score WMMAs so their latency is
// hidden behind the softmax VALU work.
__global__ void __launch_bounds__(32)
em_attn_kernel(const u16* __restrict__ Q, const u16* __restrict__ Kk,
               const u16* __restrict__ VT, const int* __restrict__ lens,
               float* __restrict__ AT) {
  __shared__ __align__(16) u16 Plds[16 * 32];
  const int lane = threadIdx.x;
  const int g = lane >> 4, ln = lane & 15;
  const int qt = blockIdx.x, h = blockIdx.y, b = blockIdx.z;
  const int bh = b * HH + h;
  const int len = lens[b];

  const u16* qrow = Q + ((size_t)bh * SS + qt * 16 + ln) * DKK;
  BFrag aq0, aq1;
  aq0.q[0] = *(const v4u*)(qrow + 16 * g);
  aq0.q[1] = *(const v4u*)(qrow + 8 + 16 * g);
  aq1.q[0] = *(const v4u*)(qrow + 32 + 16 * g);
  aq1.q[1] = *(const v4u*)(qrow + 40 + 16 * g);

  v8f o[4];
  float mrun[8], lrun[8];
  int irow[8];
#pragma unroll
  for (int v = 0; v < 8; v++) {
    o[0][v] = 0.f; o[1][v] = 0.f; o[2][v] = 0.f; o[3][v] = 0.f;
    mrun[v] = -3.4e38f; lrun[v] = 0.f;
    irow[v] = qt * 16 + v + 8 * g;
  }

  for (int jt = 0; jt < SS; jt += 32) {
    // issue V loads for this step early (consumed after softmax)
    BFrag bv[4];
#pragma unroll
    for (int t = 0; t < 4; t++) {
      const u16* vrow = VT + ((size_t)bh * DKK + t * 16 + ln) * SS + jt;
      bv[t].q[0] = *(const v4u*)(vrow + 16 * g);
      bv[t].q[1] = *(const v4u*)(vrow + 16 * g + 8);
    }

    const int j0 = jt + ln, j1 = j0 + 16;
    const u16* k0p = Kk + ((size_t)bh * SS + j0) * DKK;
    const u16* k1p = Kk + ((size_t)bh * SS + j1) * DKK;
    BFrag b0a, b0b, b1a, b1b;
    b0a.q[0] = *(const v4u*)(k0p + 16 * g);      b0a.q[1] = *(const v4u*)(k0p + 16 * g + 8);
    b0b.q[0] = *(const v4u*)(k0p + 32 + 16 * g); b0b.q[1] = *(const v4u*)(k0p + 40 + 16 * g);
    b1a.q[0] = *(const v4u*)(k1p + 16 * g);      b1a.q[1] = *(const v4u*)(k1p + 16 * g + 8);
    b1b.q[0] = *(const v4u*)(k1p + 32 + 16 * g); b1b.q[1] = *(const v4u*)(k1p + 40 + 16 * g);

    v8f c0, c1;
#pragma unroll
    for (int v = 0; v < 8; v++) { c0[v] = 0.f; c1[v] = 0.f; }
    c0 = wmma_bf16(aq0, b0a, c0);  c0 = wmma_bf16(aq1, b0b, c0);   // cols jt..jt+15
    c1 = wmma_bf16(aq0, b1a, c1);  c1 = wmma_bf16(aq1, b1b, c1);   // cols jt+16..jt+31

#pragma unroll
    for (int v = 0; v < 8; v++) {
      bool m0b = is_masked(irow[v], j0, len);
      bool m1b = is_masked(irow[v], j1, len);
      float s0 = m0b ? -3.4e38f : c0[v] * 0.125f;   // 1/sqrt(64)
      float s1 = m1b ? -3.4e38f : c1[v] * 0.125f;
      float tmax = half_max(fmaxf(s0, s1));
      float nm = fmaxf(mrun[v], tmax);
      float scale = (mrun[v] == nm) ? 1.0f : __expf(mrun[v] - nm);
      float p0 = m0b ? 0.f : __expf(s0 - nm);
      float p1 = m1b ? 0.f : __expf(s1 - nm);
      float psum = half_sum(p0 + p1);
      lrun[v] = lrun[v] * scale + psum;
      mrun[v] = nm;
      o[0][v] *= scale; o[1][v] *= scale; o[2][v] *= scale; o[3][v] *= scale;
      int pr = v + 8 * g;
      Plds[pr * 32 + ln] = f2bf(p0);
      Plds[pr * 32 + 16 + ln] = f2bf(p1);
    }
    __syncthreads();                                 // order DS writes before frag reads
    BFrag pa;                                        // remap P: D-layout -> A-layout
    pa.q[0] = *(const v4u*)(Plds + ln * 32 + 16 * g);
    pa.q[1] = *(const v4u*)(Plds + ln * 32 + 16 * g + 8);
#pragma unroll
    for (int t = 0; t < 4; t++) o[t] = wmma_bf16(pa, bv[t], o[t]);
    __syncthreads();                                 // protect Plds reuse next iteration
  }

#pragma unroll
  for (int v = 0; v < 8; v++) {
    float inv = (lrun[v] > 0.f) ? (1.0f / lrun[v]) : 0.f;
    size_t base = ((size_t)b * SS + qt * 16 + v + 8 * g) * DD + h * DKK;
    AT[base + 0 + ln]  = o[0][v] * inv;
    AT[base + 16 + ln] = o[1][v] * inv;
    AT[base + 32 + ln] = o[2][v] * inv;
    AT[base + 48 + ln] = o[3][v] * inv;
  }
}

// final: X[B,S,D] -> out[B,D,T] dropping the R_TOT right-context rows
__global__ void em_out_kernel(const float* __restrict__ X, float* __restrict__ out) {
  size_t idx = (size_t)blockIdx.x * 256 + threadIdx.x;
  if (idx >= (size_t)BB * DD * TT) return;
  int t = (int)(idx & (TT - 1));
  size_t r = idx >> 10;
  int d = (int)(r & (DD - 1));
  int b = (int)(r >> 9);
  out[idx] = X[((size_t)b * SS + RR + t) * DD + d];
}

// ---------------------------------------------------------------- launch ---
extern "C" void kernel_launch(void* const* d_in, const int* in_sizes, int n_in,
                              void* d_out, int out_size, void* d_ws, size_t ws_size,
                              hipStream_t stream) {
  (void)in_sizes; (void)n_in; (void)out_size; (void)ws_size;
  const float* audio = (const float*)d_in[0];
  const int*   alens = (const int*)d_in[1];
  const float* lw    = (const float*)d_in[2];
  const float* lb    = (const float*)d_in[3];
  const float* g_in  = (const float*)d_in[4];
  const float* b_in  = (const float*)d_in[5];
  const float* wq    = (const float*)d_in[6];
  const float* bq    = (const float*)d_in[7];
  const float* wk    = (const float*)d_in[8];
  const float* bk    = (const float*)d_in[9];
  const float* wv    = (const float*)d_in[10];
  const float* bv    = (const float*)d_in[11];
  const float* g1    = (const float*)d_in[12];
  const float* b1    = (const float*)d_in[13];
  const float* wf1   = (const float*)d_in[14];
  const float* bf1   = (const float*)d_in[15];
  const float* wf2   = (const float*)d_in[16];
  const float* bf2   = (const float*)d_in[17];
  const float* g2    = (const float*)d_in[18];
  const float* b2    = (const float*)d_in[19];

  char* ws = (char*)d_ws;
  size_t off = 0;
  auto alloc = [&](size_t bytes) -> char* {
    char* p = ws + off;
    off += (bytes + 255) & ~(size_t)255;
    return p;
  };
  int*   lens_i = (int*)alloc(64);
  u16*   WT     = (u16*)alloc((size_t)LLAY * WT_PER * 2);     // ~23 MB bf16 weights
  float* X      = (float*)alloc((size_t)MM * DD * 4);
  float* Hb     = (float*)alloc((size_t)MM * DD * 4);         // h / z
  u16*   HB     = (u16*)alloc((size_t)MM * DD * 2);           // bf16 h / y
  u16*   Qb     = (u16*)alloc((size_t)MM * DD * 2);
  u16*   Kb     = (u16*)alloc((size_t)MM * DD * 2);
  u16*   VT     = (u16*)alloc((size_t)MM * DD * 2);
  float* AT     = (float*)alloc((size_t)MM * DD * 4);         // attn / ao
  u16*   MID    = (u16*)alloc((size_t)MM * FFNN * 2);

  em_lens_kernel<<<1, 32, 0, stream>>>(alens, lens_i, (float*)d_out + (size_t)BB * DD * TT);

  // transpose+convert all weights to bf16 [N,K]
  for (int l = 0; l < LLAY; l++) {
    u16* wqT  = WT + (size_t)l * WT_PER;
    u16* wkT  = wqT + DD * DD;
    u16* wvT  = wkT + DD * DD;
    u16* wf1T = wvT + DD * DD;
    u16* wf2T = wf1T + DD * FFNN;
    int gq = (DD * DD + 255) / 256, gf = (DD * FFNN + 255) / 256;
    em_wconv_kernel<<<gq, 256, 0, stream>>>(wq  + (size_t)l * DD * DD,   wqT,  DD,   DD);
    em_wconv_kernel<<<gq, 256, 0, stream>>>(wk  + (size_t)l * DD * DD,   wkT,  DD,   DD);
    em_wconv_kernel<<<gq, 256, 0, stream>>>(wv  + (size_t)l * DD * DD,   wvT,  DD,   DD);
    em_wconv_kernel<<<gf, 256, 0, stream>>>(wf1 + (size_t)l * DD * FFNN, wf1T, DD,   FFNN);
    em_wconv_kernel<<<gf, 256, 0, stream>>>(wf2 + (size_t)l * FFNN * DD, wf2T, FFNN, DD);
  }

  em_frontend_kernel<<<MM, 128, 0, stream>>>(audio, lw, lb, X);

  for (int l = 0; l < LLAY; l++) {
    u16* wqT  = WT + (size_t)l * WT_PER;
    u16* wkT  = wqT + DD * DD;
    u16* wvT  = wkT + DD * DD;
    u16* wf1T = wvT + DD * DD;
    u16* wf2T = wf1T + DD * FFNN;

    em_ln_in_kernel<<<MM / 8, 256, 0, stream>>>(X, g_in + l * DD, b_in + l * DD, Hb, HB);

    dim3 gqkv(MM / 128, DD / 64);
    em_gemm128x64<<<gqkv, 256, 0, stream>>>(HB, wqT, bq + l * DD, Qb, DD, DD, 2);
    em_gemm128x64<<<gqkv, 256, 0, stream>>>(HB, wkT, bk + l * DD, Kb, DD, DD, 2);
    em_gemm128x64<<<gqkv, 256, 0, stream>>>(HB, wvT, bv + l * DD, VT, DD, DD, 3);

    em_attn_kernel<<<dim3(SS / 16, HH, BB), 32, 0, stream>>>(Qb, Kb, VT, lens_i, AT);

    em_ln_res_kernel<<<MM / 8, 256, 0, stream>>>(AT, Hb, g1 + l * DD, b1 + l * DD, HB);

    em_gemm128x64<<<dim3(MM / 128, FFNN / 64), 256, 0, stream>>>(HB, wf1T, bf1 + l * FFNN,
                                                                 MID, DD, FFNN, 1);
    em_gemm128x64<<<gqkv, 256, 0, stream>>>(MID, wf2T, bf2 + l * DD, Hb, FFNN, DD, 0);

    em_ln_res2_kernel<<<MM / 8, 256, 0, stream>>>(Hb, AT, g2 + l * DD, b2 + l * DD, X);
  }

  em_out_kernel<<<(BB * DD * TT) / 256, 256, 0, stream>>>(X, (float*)d_out);
}
